// LSTMEncoderDecoder_47167330844839
// MI455X (gfx1250) — matrix operand order repaired and checked by reading, not compiled, not run on hardware
//
#include <hip/hip_runtime.h>

// ---------------------------------------------------------------------------
// LSTM encoder-decoder for MI455X (gfx1250), wave32 + v_wmma_f32_16x16x32_f16.
// One persistent workgroup per 32-sample batch tile runs the whole sequence.
// Wave w owns gate columns {g*256 + 16w .. +15} for all 4 gates -> the
// i/f/g/o recombination and c-state update are fully register-local.
// ---------------------------------------------------------------------------

typedef __attribute__((ext_vector_type(16))) _Float16 v16h;
typedef __attribute__((ext_vector_type(8)))  _Float16 v8h;
typedef __attribute__((ext_vector_type(8)))  float    v8f;

#define BATCH   8192
#define TSTEPS  11
#define FIN     10
#define HID     256
#define DSTEPS  80
#define POUT    2

#define MTILE   32      // samples per workgroup
#define NMT     2       // 16-row M tiles per workgroup
#define BLOCK   512     // 16 waves; wave w owns hidden columns [16w, 16w+16)
#define HSTR    264     // LDS h row stride (elements): 256 + 8 pad (bank step 4)
#define XSTR    40      // LDS x row stride (elements): 32 + 8 pad
#define XPAD    32      // K-padded input-feature count

// ---- fp16 workspace layout (element offsets) ------------------------------
#define O_EIH0  0                               // enc Wih0 padded [1024 x 32]
#define O_EHH0  (O_EIH0 + 1024*32)              // enc Whh0        [1024 x 256]
#define O_EIH1  (O_EHH0 + 1024*256)             // enc Wih1        [1024 x 256]
#define O_EHH1  (O_EIH1 + 1024*256)             // enc Whh1        [1024 x 256]
#define O_DIH0  (O_EHH1 + 1024*256)             // dec Wih0 padded [1024 x 32]
#define O_DHH0  (O_DIH0 + 1024*32)              // dec Whh0        [1024 x 256]
#define O_DIH1  (O_DHH0 + 1024*256)             // dec Wih1        [1024 x 256]
#define O_DHH1  (O_DIH1 + 1024*256)             // dec Whh1        [1024 x 256]

__device__ __forceinline__ float sigmf(float x) {
    return 1.0f / (1.0f + __expf(-x));
}
__device__ __forceinline__ float tanh_fast(float x) {
    // stable for large |x|: exp->inf gives 1, exp->0 gives -1
    return 1.0f - 2.0f / (1.0f + __expf(2.0f * x));
}

// Accumulate gates[m, g*256 + jw + 0..15] += Asrc[m, 0..kx) * W[n, 0..kx)^T
// A fragment (16x32 f16): lane row = ln16; VGPR[0:7] K = k0+16*half+[0..15]
// contiguous, same k order as B   -> two 16B LDS loads.
// B fragment (32x16 f16): lane col = ln16; K = k0+16*half+[0..15] contiguous
// -> weight row n, one 32B global read (weights stay L2-resident).
__device__ __forceinline__ void mma_k(v8f acc[NMT][4],
                                      const _Float16* a, int astride, int kx,
                                      const _Float16* __restrict__ w,
                                      int half, int ln16, int jw)
{
#pragma unroll 2
    for (int k0 = 0; k0 < kx; k0 += 32) {
        v16h A[NMT];
#pragma unroll
        for (int mt = 0; mt < NMT; ++mt) {
            const _Float16* ap = a + (mt * 16 + ln16) * astride + k0 + half * 16;
            v8h lo = *(const v8h*)ap;
            v8h hi = *(const v8h*)(ap + 8);
            A[mt] = __builtin_shufflevector(lo, hi,
                        0,1,2,3,4,5,6,7,8,9,10,11,12,13,14,15);
        }
#pragma unroll
        for (int g = 0; g < 4; ++g) {
            v16h Bv = *(const v16h*)(w + (size_t)(g * 256 + jw + ln16) * kx
                                       + k0 + half * 16);
#pragma unroll
            for (int mt = 0; mt < NMT; ++mt) {
                acc[mt][g] = __builtin_amdgcn_wmma_f32_16x16x32_f16(
                    false, A[mt], false, Bv, (short)0, acc[mt][g], false, false);
            }
        }
    }
}

// One LSTM cell step for the whole batch tile.
// gates = xsrc @ Wih^T + hbuf @ Whh^T + b ; update c (regs) and h (LDS).
__device__ __forceinline__ void lstm_layer(
    const _Float16* xsrc, int xstride, int kx,
    const _Float16* __restrict__ wih,
    _Float16* hbuf,
    const _Float16* __restrict__ whh,
    const float4 bias, float creg[NMT][8],
    int half, int ln16, int jw)
{
    v8f acc[NMT][4] = {};
    mma_k(acc, xsrc, xstride, kx, wih, half, ln16, jw);
    mma_k(acc, hbuf, HSTR,   HID, whh, half, ln16, jw);
    __syncthreads();  // all waves done reading old h before anyone rewrites it
#pragma unroll
    for (int mt = 0; mt < NMT; ++mt) {
#pragma unroll
        for (int r = 0; r < 8; ++r) {
            // C/D layout: VGPR r -> row m = mt*16 + 8*half + r, col = jw+ln16
            float iv = acc[mt][0][r] + bias.x;
            float fv = acc[mt][1][r] + bias.y;
            float gv = acc[mt][2][r] + bias.z;
            float ov = acc[mt][3][r] + bias.w;
            float cn = sigmf(fv) * creg[mt][r] + sigmf(iv) * tanh_fast(gv);
            float hn = sigmf(ov) * tanh_fast(cn);
            creg[mt][r] = cn;
            int m = mt * 16 + half * 8 + r;
            hbuf[m * HSTR + jw + ln16] = (_Float16)hn;
        }
    }
    __syncthreads();
}

__global__ __launch_bounds__(BLOCK, 1)
void lstm_encdec_kernel(
    const float* __restrict__ x,
    const _Float16* __restrict__ ws,
    const float* __restrict__ enc_b0, const float* __restrict__ enc_b1,
    const float* __restrict__ dec_b0, const float* __restrict__ dec_b1,
    const float* __restrict__ Wlin,   const float* __restrict__ blin,
    float* __restrict__ out)
{
    __shared__ __align__(16) _Float16 h0buf[MTILE * HSTR];
    __shared__ __align__(16) _Float16 h1buf[MTILE * HSTR];
    __shared__ __align__(16) _Float16 xbuf [MTILE * XSTR];

    const int tid  = threadIdx.x;
    const int wave = tid >> 5;
    const int lane = tid & 31;
    const int half = lane >> 4;
    const int ln16 = lane & 15;
    const int jw   = wave * 16;        // hidden-unit column block this wave owns
    const int b0   = blockIdx.x * MTILE;
    const int col  = jw + ln16;

    const _Float16* wEih0 = ws + O_EIH0;
    const _Float16* wEhh0 = ws + O_EHH0;
    const _Float16* wEih1 = ws + O_EIH1;
    const _Float16* wEhh1 = ws + O_EHH1;
    const _Float16* wDih0 = ws + O_DIH0;
    const _Float16* wDhh0 = ws + O_DHH0;
    const _Float16* wDih1 = ws + O_DIH1;
    const _Float16* wDhh1 = ws + O_DHH1;

    // per-lane gate biases (i,f,g,o) for each of the 4 LSTM cells
    const float4 bE0 = make_float4(enc_b0[col], enc_b0[256+col], enc_b0[512+col], enc_b0[768+col]);
    const float4 bE1 = make_float4(enc_b1[col], enc_b1[256+col], enc_b1[512+col], enc_b1[768+col]);
    const float4 bD0 = make_float4(dec_b0[col], dec_b0[256+col], dec_b0[512+col], dec_b0[768+col]);
    const float4 bD1 = make_float4(dec_b1[col], dec_b1[256+col], dec_b1[512+col], dec_b1[768+col]);

    float c0[NMT][8] = {};
    float c1[NMT][8] = {};

    for (int i = tid; i < MTILE * HSTR; i += BLOCK) {
        h0buf[i] = (_Float16)0.f;
        h1buf[i] = (_Float16)0.f;
    }
    for (int i = tid; i < MTILE * XSTR; i += BLOCK) xbuf[i] = (_Float16)0.f;
    __syncthreads();

    // ---------------- encoder: 11 steps, 2 layers -------------------------
    for (int t = 0; t < TSTEPS; ++t) {
        for (int i = tid; i < MTILE * FIN; i += BLOCK) {
            int m = i / FIN, f = i % FIN;
            xbuf[m * XSTR + f] =
                (_Float16)x[(size_t)(b0 + m) * (TSTEPS * FIN) + t * FIN + f];
        }
        __syncthreads();
        lstm_layer(xbuf,  XSTR, XPAD, wEih0, h0buf, wEhh0, bE0, c0, half, ln16, jw);
        lstm_layer(h0buf, HSTR, HID,  wEih1, h1buf, wEhh1, bE1, c1, half, ln16, jw);
    }

    // ---------------- decoder init: x_in = x[:, -1, :P] -------------------
    for (int i = tid; i < MTILE * XSTR; i += BLOCK) xbuf[i] = (_Float16)0.f;
    __syncthreads();
    for (int i = tid; i < MTILE * POUT; i += BLOCK) {
        int m = i >> 1, p = i & 1;
        xbuf[m * XSTR + p] =
            (_Float16)x[(size_t)(b0 + m) * (TSTEPS * FIN) + (TSTEPS - 1) * FIN + p];
    }
    __syncthreads();

    // ---------------- decoder: 80 steps, 2 layers + linear feedback -------
    for (int sd = 0; sd < DSTEPS; ++sd) {
        lstm_layer(xbuf,  XSTR, XPAD, wDih0, h0buf, wDhh0, bD0, c0, half, ln16, jw);
        lstm_layer(h0buf, HSTR, HID,  wDih1, h1buf, wDhh1, bD1, c1, half, ln16, jw);

        if (tid < MTILE * POUT) {            // 64 lanes: out = h1 @ Wlin^T + b
            int m = tid >> 1, p = tid & 1;
            float s = blin[p];
            const _Float16* hr = h1buf + m * HSTR;
            const float*    wl = Wlin + p * HID;
            for (int j = 0; j < HID; ++j) s += (float)hr[j] * wl[j];
            out[(size_t)(b0 + m) * (DSTEPS * POUT) + sd * POUT + p] = s;
            xbuf[m * XSTR + p] = (_Float16)s;   // feedback input for next step
        }
        __syncthreads();
    }
}

// fp32 -> fp16 weight conversion with K-padding (rows kept, cols padded w/ 0)
__global__ void convert_pad_kernel(const float* __restrict__ src,
                                   _Float16* __restrict__ dst,
                                   int rows, int scols, int dcols)
{
    int idx = blockIdx.x * blockDim.x + threadIdx.x;
    if (idx >= rows * dcols) return;
    int r = idx / dcols, c = idx % dcols;
    dst[idx] = (c < scols) ? (_Float16)src[r * scols + c] : (_Float16)0.f;
}

extern "C" void kernel_launch(void* const* d_in, const int* in_sizes, int n_in,
                              void* d_out, int out_size, void* d_ws, size_t ws_size,
                              hipStream_t stream)
{
    // setup_inputs() dict order:
    // 0:x  1:enc_Wih0 2:enc_Whh0 3:enc_b0 4:dec_Wih0 5:dec_Whh0 6:dec_b0
    //      7:enc_Wih1 8:enc_Whh1 9:enc_b1 10:dec_Wih1 11:dec_Whh1 12:dec_b1
    // 13:Wlin 14:blin
    const float* x      = (const float*)d_in[0];
    const float* eWih0  = (const float*)d_in[1];
    const float* eWhh0  = (const float*)d_in[2];
    const float* eb0    = (const float*)d_in[3];
    const float* dWih0  = (const float*)d_in[4];
    const float* dWhh0  = (const float*)d_in[5];
    const float* db0    = (const float*)d_in[6];
    const float* eWih1  = (const float*)d_in[7];
    const float* eWhh1  = (const float*)d_in[8];
    const float* eb1    = (const float*)d_in[9];
    const float* dWih1  = (const float*)d_in[10];
    const float* dWhh1  = (const float*)d_in[11];
    const float* db1    = (const float*)d_in[12];
    const float* Wlin   = (const float*)d_in[13];
    const float* blin   = (const float*)d_in[14];
    float* out          = (float*)d_out;
    _Float16* ws        = (_Float16*)d_ws;

    const int thr = 256;
    auto blks = [](int n, int t) { return (n + t - 1) / t; };

    convert_pad_kernel<<<blks(1024*32,  thr), thr, 0, stream>>>(eWih0, ws + O_EIH0, 1024, FIN, XPAD);
    convert_pad_kernel<<<blks(1024*256, thr), thr, 0, stream>>>(eWhh0, ws + O_EHH0, 1024, HID, HID);
    convert_pad_kernel<<<blks(1024*256, thr), thr, 0, stream>>>(eWih1, ws + O_EIH1, 1024, HID, HID);
    convert_pad_kernel<<<blks(1024*256, thr), thr, 0, stream>>>(eWhh1, ws + O_EHH1, 1024, HID, HID);
    convert_pad_kernel<<<blks(1024*32,  thr), thr, 0, stream>>>(dWih0, ws + O_DIH0, 1024, POUT, XPAD);
    convert_pad_kernel<<<blks(1024*256, thr), thr, 0, stream>>>(dWhh0, ws + O_DHH0, 1024, HID, HID);
    convert_pad_kernel<<<blks(1024*256, thr), thr, 0, stream>>>(dWih1, ws + O_DIH1, 1024, HID, HID);
    convert_pad_kernel<<<blks(1024*256, thr), thr, 0, stream>>>(dWhh1, ws + O_DHH1, 1024, HID, HID);

    lstm_encdec_kernel<<<BATCH / MTILE, BLOCK, 0, stream>>>(
        x, ws, eb0, eb1, db0, db1, Wlin, blin, out);
}
